// HyperSpace_36120674959667
// MI455X (gfx1250) — matrix operand order repaired and checked by reading, not compiled, not run on hardware
//
#include <hip/hip_runtime.h>
#include <hip/hip_bf16.h>

typedef __attribute__((ext_vector_type(16))) __bf16 v16bf;
typedef __attribute__((ext_vector_type(8)))  __bf16 v8bf;
typedef __attribute__((ext_vector_type(8)))  float  v8f;

#define B_ROWS 16384
#define F_DIM  256
#define M_BINS 64
#define D_DIRS 4096
#define EPSF   1e-5f
#define NBINS  1024

#define ROWS_PER_BLOCK 256   // 8 waves x 2 tiles x 16 rows
#define NSPLITS        8
#define COLS_PER_SPLIT (D_DIRS / NSPLITS)    // 512
#define CHUNKS         (COLS_PER_SPLIT / 16) // 32

// ---------------------------------------------------------------------------
// 1) Standardize rows, emit bf16 copy, row L2 norm, digitize into magnitude bin
// ---------------------------------------------------------------------------
__global__ void normalize_rows(const float* __restrict__ vec,
                               const float* __restrict__ mean,
                               const float* __restrict__ stdv,
                               const float* __restrict__ refmag,
                               __bf16* __restrict__ v_bf,
                               int* __restrict__ mag_idx) {
    __shared__ float red[256];
    const int row = blockIdx.x;
    const int f   = threadIdx.x;
    float x  = vec[(size_t)row * F_DIM + f];
    float vn = (x - mean[f]) / (stdv[f] + EPSF);
    v_bf[(size_t)row * F_DIM + f] = (__bf16)vn;
    red[f] = vn * vn;
    __syncthreads();
    for (int s = 128; s > 0; s >>= 1) {
        if (f < s) red[f] += red[f + s];
        __syncthreads();
    }
    if (f == 0) {
        float nrm = sqrtf(red[0]);
        int lo = 0, hi = M_BINS;
        while (lo < hi) {
            int mid = (lo + hi) >> 1;
            if (refmag[mid] <= nrm) lo = mid + 1; else hi = mid;
        }
        mag_idx[row] = (lo < M_BINS - 1) ? lo : (M_BINS - 1);
    }
}

// ---------------------------------------------------------------------------
// 2) Reference directions -> bf16
// ---------------------------------------------------------------------------
__global__ void dirs_to_bf16(const float* __restrict__ dirs,
                             __bf16* __restrict__ out, int n) {
    int i = blockIdx.x * blockDim.x + threadIdx.x;
    if (i < n) out[i] = (__bf16)dirs[i];
}

// ---------------------------------------------------------------------------
// 3) Rank machinery: 1024-bin weighted histogram + inclusive scan.
//    P[c] = sum of all count entries <= c  (== cum[searchsorted-1] in ref)
// ---------------------------------------------------------------------------
__global__ void zero_ws(unsigned int* __restrict__ whist,
                        unsigned long long* __restrict__ packed) {
    int i = blockIdx.x * blockDim.x + threadIdx.x;
    if (i < NBINS) whist[i] = 0u;
    if (i < B_ROWS) packed[i] = 0ull;
}

__global__ void hist_kernel(const int* __restrict__ counts,
                            unsigned int* __restrict__ whist, int n) {
    int i = blockIdx.x * blockDim.x + threadIdx.x;
    if (i < n) {
        int c = counts[i];
        int b = c < 0 ? 0 : (c > NBINS - 1 ? NBINS - 1 : c);
        if (b > 0) atomicAdd(&whist[b], (unsigned int)b);
    }
}

__global__ void scan_kernel(const unsigned int* __restrict__ whist,
                            float* __restrict__ P) {
    __shared__ unsigned int s[NBINS];
    int t = threadIdx.x;
    s[t] = whist[t];
    __syncthreads();
    for (int off = 1; off < NBINS; off <<= 1) {
        unsigned int v = (t >= off) ? s[t - off] : 0u;
        __syncthreads();
        s[t] += v;
        __syncthreads();
    }
    P[t] = (float)s[t];
    if (t == NBINS - 1) P[NBINS] = (float)(s[t] > 0u ? s[t] : 1u);
}

// ---------------------------------------------------------------------------
// 4) WMMA GEMM + argmax. Block = 8 waves x 32 rows = 256 rows; grid.y splits D.
//    All waves consume the same 16 columns per chunk (WGP$ reuse); each B
//    fragment feeds two WMMAs (register reuse); B loads flow through a 4-deep
//    ring so 4 loads stay in flight behind the WMMA pipe. Cross-split argmax
//    merges via atomicMax on packed (orderable score bits << 32 | ~index).
// ---------------------------------------------------------------------------
__device__ __forceinline__ unsigned int fenc(float f) {
    unsigned int u = __float_as_uint(f);
    return (u & 0x80000000u) ? ~u : (u | 0x80000000u);
}

__global__ void __launch_bounds__(256)
gemm_argmax(const __bf16* __restrict__ v_bf,
            const __bf16* __restrict__ dirs_bf,
            unsigned long long* __restrict__ packed) {
    const int lane = threadIdx.x & 31;
    const int wave = threadIdx.x >> 5;
    const int g    = lane >> 4;    // lane half: K sub-range select
    const int lr   = lane & 15;    // row (A) / col (B) within tile

    const int rowBase0 = blockIdx.x * ROWS_PER_BLOCK + wave * 32;
    const int rowBase1 = rowBase0 + 16;
    const int colSplit = blockIdx.y * COLS_PER_SPLIT;

    // Hoist both A tiles (K = 0..255) into registers.
    // A k order used here (matches B): lane l holds row l%16; element e holds
    // K = (l/16)*16 + e  -> two contiguous 8-half loads.
    const __bf16* aRow0 = v_bf + (size_t)(rowBase0 + lr) * F_DIM;
    const __bf16* aRow1 = v_bf + (size_t)(rowBase1 + lr) * F_DIM;
    v16bf A0[8], A1[8];
#pragma unroll
    for (int ks = 0; ks < 8; ++ks) {
        const int k0 = ks * 32;
        v8bf lo0 = *(const v8bf*)(aRow0 + k0 + g * 16);
        v8bf hi0 = *(const v8bf*)(aRow0 + k0 + g * 16 + 8);
        v8bf lo1 = *(const v8bf*)(aRow1 + k0 + g * 16);
        v8bf hi1 = *(const v8bf*)(aRow1 + k0 + g * 16 + 8);
#pragma unroll
        for (int e = 0; e < 8; ++e) {
            A0[ks][e] = lo0[e]; A0[ks][e + 8] = hi0[e];
            A1[ks][e] = lo1[e]; A1[ks][e + 8] = hi1[e];
        }
    }

    float best0[8], best1[8];
    int   bidx0[8], bidx1[8];
#pragma unroll
    for (int j = 0; j < 8; ++j) {
        best0[j] = -3.0e38f; bidx0[j] = 0;
        best1[j] = -3.0e38f; bidx1[j] = 0;
    }

    for (int chunk = 0; chunk < CHUNKS; ++chunk) {
        const int colBase = colSplit + chunk * 16;
        // B k order used here (matches A): lane l holds col l%16, K = (l/16)*16 + e
        const __bf16* bCol = dirs_bf + (size_t)(colBase + lr) * F_DIM + g * 16;
        __builtin_prefetch(bCol + 16 * F_DIM, 0, 1);  // warm next chunk's B line

        // 4-deep B ring: fill 4 slots, then each step consumes slot ks&3 and
        // immediately refills it with fragment ks+4 (keeps 4 loads in flight).
        v16bf Bbuf[4];
#pragma unroll
        for (int p = 0; p < 4; ++p)
            Bbuf[p] = *(const v16bf*)(bCol + p * 32);

        v8f c0 = {}, c1 = {};
#pragma unroll
        for (int ks = 0; ks < 8; ++ks) {
            c0 = __builtin_amdgcn_wmma_f32_16x16x32_bf16(
                     false, A0[ks], false, Bbuf[ks & 3], (short)0, c0, false, false);
            c1 = __builtin_amdgcn_wmma_f32_16x16x32_bf16(
                     false, A1[ks], false, Bbuf[ks & 3], (short)0, c1, false, false);
            if (ks + 4 < 8)
                Bbuf[ks & 3] = *(const v16bf*)(bCol + (ks + 4) * 32);
        }
        const int col = colBase + lr;
#pragma unroll
        for (int j = 0; j < 8; ++j) {
            if (c0[j] > best0[j]) { best0[j] = c0[j]; bidx0[j] = col; }
            if (c1[j] > best1[j]) { best1[j] = c1[j]; bidx1[j] = col; }
        }
    }

    // Reduce across the 16 lanes sharing each row; commit via atomicMax.
#pragma unroll
    for (int j = 0; j < 8; ++j) {
        float bv = best0[j]; int bi = bidx0[j];
#pragma unroll
        for (int m = 8; m >= 1; m >>= 1) {
            float ov = __shfl_xor(bv, m, 32);
            int   oi = __shfl_xor(bi, m, 32);
            if (ov > bv || (ov == bv && oi < bi)) { bv = ov; bi = oi; }
        }
        if (lr == 0) {
            unsigned long long p =
                ((unsigned long long)fenc(bv) << 32) | (unsigned int)(~bi);
            atomicMax(&packed[rowBase0 + g * 8 + j], p);
        }
    }
#pragma unroll
    for (int j = 0; j < 8; ++j) {
        float bv = best1[j]; int bi = bidx1[j];
#pragma unroll
        for (int m = 8; m >= 1; m >>= 1) {
            float ov = __shfl_xor(bv, m, 32);
            int   oi = __shfl_xor(bi, m, 32);
            if (ov > bv || (ov == bv && oi < bi)) { bv = ov; bi = oi; }
        }
        if (lr == 0) {
            unsigned long long p =
                ((unsigned long long)fenc(bv) << 32) | (unsigned int)(~bi);
            atomicMax(&packed[rowBase1 + g * 8 + j], p);
        }
    }
}

// ---------------------------------------------------------------------------
// 5) Finalize: decode argmax, gather count, probability, rank.
// ---------------------------------------------------------------------------
__global__ void finalize(const unsigned long long* __restrict__ packed,
                         const int*   __restrict__ mag_idx,
                         const int*   __restrict__ counts,
                         const float* __restrict__ P,
                         float* __restrict__ out) {
    int row = blockIdx.x * blockDim.x + threadIdx.x;
    if (row < B_ROWS) {
        unsigned long long p = packed[row];
        int bi = (int)(~(unsigned int)(p & 0xFFFFFFFFull));
        int mi = mag_idx[row];
        int c  = counts[(size_t)mi * D_DIRS + bi];
        int cb = c < 0 ? 0 : (c > NBINS - 1 ? NBINS - 1 : c);
        float total = P[NBINS];
        float cum   = P[cb];
        out[row]              = (float)c;
        out[B_ROWS + row]     = (float)c / total;
        out[2 * B_ROWS + row] = (cum - 0.5f * (float)c) / total;
    }
}

// ---------------------------------------------------------------------------
extern "C" void kernel_launch(void* const* d_in, const int* in_sizes, int n_in,
                              void* d_out, int out_size, void* d_ws, size_t ws_size,
                              hipStream_t stream) {
    (void)in_sizes; (void)n_in; (void)out_size; (void)ws_size;
    const float* vectors = (const float*)d_in[0];
    const float* mean    = (const float*)d_in[1];
    const float* stdv    = (const float*)d_in[2];
    const float* refmag  = (const float*)d_in[3];
    const float* dirs    = (const float*)d_in[4];
    const int*   counts  = (const int*)d_in[5];
    float*       out     = (float*)d_out;

    char* ws = (char*)d_ws;
    __bf16* v_bf    = (__bf16*)ws;           ws += (size_t)B_ROWS * F_DIM * 2;  // 8 MB
    __bf16* dirs_bf = (__bf16*)ws;           ws += (size_t)D_DIRS * F_DIM * 2;  // 2 MB
    int*    mag_idx = (int*)ws;              ws += (size_t)B_ROWS * 4;
    unsigned int* whist = (unsigned int*)ws; ws += (size_t)NBINS * 4;
    float*  P       = (float*)ws;            ws += (size_t)(NBINS + 1) * 4 + 4; // pad to 8B
    unsigned long long* packed = (unsigned long long*)ws;
    ws += (size_t)B_ROWS * 8;                                                   // 128 KB

    normalize_rows<<<B_ROWS, 256, 0, stream>>>(vectors, mean, stdv, refmag, v_bf, mag_idx);
    dirs_to_bf16<<<(D_DIRS * F_DIM) / 256, 256, 0, stream>>>(dirs, dirs_bf, D_DIRS * F_DIM);
    zero_ws<<<B_ROWS / 256, 256, 0, stream>>>(whist, packed);
    hist_kernel<<<(M_BINS * D_DIRS) / 256, 256, 0, stream>>>(counts, whist, M_BINS * D_DIRS);
    scan_kernel<<<1, NBINS, 0, stream>>>(whist, P);
    gemm_argmax<<<dim3(B_ROWS / ROWS_PER_BLOCK, NSPLITS), 256, 0, stream>>>(v_bf, dirs_bf, packed);
    finalize<<<B_ROWS / 256, 256, 0, stream>>>(packed, mag_idx, counts, P, out);
}
